// Former_10574209483218
// MI455X (gfx1250) — compile-verified, matrix operand order checked by reading, not run on hardware
//
#include <hip/hip_runtime.h>

// ---------------- problem constants ----------------
#define BATCH   4
#define SEQ     2048
#define TOK     (BATCH * SEQ)      // 8192
#define DMODEL  1024
#define NHEADS  16
#define DPH     64                 // d per head
#define DE      2048               // MLP expand
#define DQVK    3072

typedef __bf16 bf16;
typedef __attribute__((ext_vector_type(16))) __bf16 v16bf;
typedef __attribute__((ext_vector_type(8)))  float  v8f;

union Frag {
  v16bf v;
  bf16  e[16];
  uint4 q[2];
};

// ---------------- elementwise converts ----------------
__global__ __launch_bounds__(256) void cvt_bf16(const float* __restrict__ src,
                                                bf16* __restrict__ dst, size_t n) {
  size_t i = (size_t)blockIdx.x * 256 + threadIdx.x;
  if (i < n) dst[i] = (bf16)src[i];
}

// src is [K][N] row-major fp32; dst is [N][K] row-major bf16 (pre-transposed weights)
__global__ __launch_bounds__(256) void cvt_transpose_bf16(const float* __restrict__ src,
                                                          bf16* __restrict__ dst,
                                                          int K, int N) {
  size_t i = (size_t)blockIdx.x * 256 + threadIdx.x;
  if (i >= (size_t)K * N) return;
  int k = (int)(i / N);
  int n = (int)(i % N);
  dst[(size_t)n * K + k] = (bf16)src[i];
}

// ---------------- GEMM: out = relu6(A[M,K] @ W[K,N] + bias) ----------------
// A: bf16 [M,*] row stride lda; WT: bf16 [N][K] (pre-transposed)
// Block tile 128(M) x 128(N), K-step 32, double-buffered async-to-LDS staging.
// 8 waves; wave tile 32(M) x 64(N) -> 8 WMMAs per K-step per wave.
// LDS rows padded 64B->80B (word stride 20 -> conflict-free across 16 lanes).
// Dynamic LDS layout (bytes): Abuf0 [0,10240) Abuf1 [10240,20480)
//                             Bbuf0 [20480,30720) Bbuf1 [30720,40960)
#define GROW_E   40      // padded row stride in elements (80 bytes)
#define GBUF_B   10240   // one buffer: 128 rows * 80 B
#define GSMEM_BYTES 40960

__global__ __launch_bounds__(256) void gemm_bias_relu6(
    const bf16* __restrict__ A, int lda,
    const bf16* __restrict__ WT,
    const float* __restrict__ bias,
    float* __restrict__ outF, bf16* __restrict__ outB,
    int M, int N, int K) {
  extern __shared__ char smem[];
  const int tid  = threadIdx.x;
  const int wave = tid >> 5;
  const int lane = tid & 31;
  const int wm   = wave >> 1;          // 0..3 (32-row group)
  const int wn   = wave & 1;           // 0..1 (64-col group)
  const int bm   = blockIdx.y * 128;
  const int bn   = blockIdx.x * 128;

  const int m_f  = lane & 15;
  const int koff = (lane >> 4) * 8;

  // async staging: thread t fills row r = t>>1 (0..127),
  // 32 contiguous bytes at element offset (t&1)*16 (two b128 transfers, off 0/16)
  const int r_ld = tid >> 1;
  const int c_ld = (tid & 1) * 16;     // element offset within 32-k row
  unsigned long long ga = (unsigned long long)(uintptr_t)(A  + (size_t)(bm + r_ld) * lda + c_ld);
  unsigned long long gb = (unsigned long long)(uintptr_t)(WT + (size_t)(bn + r_ld) * K   + c_ld);
  const unsigned ldsA_row = (unsigned)(r_ld * 80 + c_ld * 2);            // bytes
  const unsigned ldsB_row = (unsigned)(2 * GBUF_B + r_ld * 80 + c_ld * 2);

  auto issue_tile = [&](int buf, unsigned long long a64, unsigned long long b64) {
    unsigned la = ldsA_row + (unsigned)buf * GBUF_B;
    unsigned lb = ldsB_row + (unsigned)buf * GBUF_B;
    asm volatile(
        "global_load_async_to_lds_b128 %0, %2, off\n\t"
        "global_load_async_to_lds_b128 %0, %2, off offset:16\n\t"
        "global_load_async_to_lds_b128 %1, %3, off\n\t"
        "global_load_async_to_lds_b128 %1, %3, off offset:16"
        :: "v"(la), "v"(lb), "v"(a64), "v"(b64)
        : "memory");
  };

  v8f acc[2][4];
#pragma unroll
  for (int i = 0; i < 2; ++i)
#pragma unroll
    for (int j = 0; j < 4; ++j) acc[i][j] = v8f{0};

  auto compute = [&](int buf) {
    const bf16* sAb = (const bf16*)(smem + buf * GBUF_B);
    const bf16* sBb = (const bf16*)(smem + 2 * GBUF_B + buf * GBUF_B);
    Frag af[2];
#pragma unroll
    for (int i = 0; i < 2; ++i) {
      int row = wm * 32 + i * 16 + m_f;
      af[i].q[0] = *(const uint4*)(sAb + row * GROW_E + koff);
      af[i].q[1] = *(const uint4*)(sAb + row * GROW_E + 16 + koff);
    }
#pragma unroll
    for (int j = 0; j < 4; ++j) {
      Frag bfr;
      int col = wn * 64 + j * 16 + m_f;
      bfr.q[0] = *(const uint4*)(sBb + col * GROW_E + koff);
      bfr.q[1] = *(const uint4*)(sBb + col * GROW_E + 16 + koff);
      acc[0][j] = __builtin_amdgcn_wmma_f32_16x16x32_bf16(
          false, af[0].v, false, bfr.v, (short)0, acc[0][j], false, false);
      acc[1][j] = __builtin_amdgcn_wmma_f32_16x16x32_bf16(
          false, af[1].v, false, bfr.v, (short)0, acc[1][j], false, false);
    }
  };

  const int nk = K / 32;
  issue_tile(0, ga, gb);
  ga += 64; gb += 64;

  // steady state: branch-free body (issue next, wait current, compute)
  for (int it = 0; it < nk - 1; ++it) {
    issue_tile((it & 1) ^ 1, ga, gb);
    ga += 64; gb += 64;
    asm volatile("s_wait_asynccnt 0x4" ::: "memory");  // current buffer's 4 landed
    __syncthreads();                                   // all waves' portions visible
    compute(it & 1);
    __syncthreads();                                   // reads done before refill
  }
  // peeled last iteration
  asm volatile("s_wait_asynccnt 0x0" ::: "memory");
  __syncthreads();
  compute((nk - 1) & 1);

  const int mbase = (lane >> 4) * 8;
  const int nloc  = lane & 15;
#pragma unroll
  for (int i = 0; i < 2; ++i)
#pragma unroll
    for (int j = 0; j < 4; ++j) {
      int gn  = bn + wn * 64 + j * 16 + nloc;
      float b = bias ? bias[gn] : 0.0f;
#pragma unroll
      for (int r = 0; r < 8; ++r) {
        int gm  = bm + wm * 32 + i * 16 + mbase + r;
        float v = acc[i][j][r] + b;
        v = fminf(fmaxf(v, 0.0f), 6.0f);
        if (outF) outF[(size_t)gm * N + gn] = v;
        if (outB) outB[(size_t)gm * N + gn] = (bf16)v;
      }
    }
}

// ---------------- flash attention ----------------
// Q/K/V: bf16 [B*SEQ, DMODEL], head-interleaved: channel = d*NHEADS + h
// O:     bf16 [B*SEQ, DMODEL], head-blocked:    channel = h*DPH + d
__device__ inline float half_max(float v) {
  v = fmaxf(v, __shfl_xor(v, 1));
  v = fmaxf(v, __shfl_xor(v, 2));
  v = fmaxf(v, __shfl_xor(v, 4));
  v = fmaxf(v, __shfl_xor(v, 8));
  return v;
}
__device__ inline float half_sum(float v) {
  v += __shfl_xor(v, 1);
  v += __shfl_xor(v, 2);
  v += __shfl_xor(v, 4);
  v += __shfl_xor(v, 8);
  return v;
}

__global__ __launch_bounds__(128) void flash_attn(
    const bf16* __restrict__ Qb, const bf16* __restrict__ Kb,
    const bf16* __restrict__ Vb, bf16* __restrict__ Ob, float inv_scale) {
  const int qblk = blockIdx.x;   // 64-row q block
  const int h    = blockIdx.y;
  const int b    = blockIdx.z;
  const int tid  = threadIdx.x;
  const int wave = tid >> 5;     // 0..3, 16 q rows each
  const int lane = tid & 31;

  // rows padded to multiples of 16 B with conflict-free word strides
  __shared__ bf16 sK[32][72];       // [token][d]   144 B rows
  __shared__ bf16 sVt[64][40];      // [d][token]    80 B rows
  __shared__ bf16 sP[4][16][40];    // per-wave P scratch, 80 B rows

  const size_t base = (size_t)b * SEQ;
  const int m_f  = lane & 15;
  const int koff = (lane >> 4) * 8;
  const int mbase = (lane >> 4) * 8;
  const int nloc  = lane & 15;

  // Q fragments for this wave's 16 rows (K-dim = 64 -> 2 chunks of 32)
  Frag qf[2];
  {
    const int qtok = qblk * 64 + wave * 16 + m_f;
    const bf16* qrow = Qb + (base + qtok) * DMODEL;
#pragma unroll
    for (int c = 0; c < 2; ++c)
#pragma unroll
      for (int i = 0; i < 8; ++i) {
        qf[c].e[i]     = qrow[(c * 32 + koff + i) * NHEADS + h];
        qf[c].e[8 + i] = qrow[(c * 32 + 16 + koff + i) * NHEADS + h];
      }
  }

  v8f oacc[4] = {v8f{0}, v8f{0}, v8f{0}, v8f{0}};
  float rmax[8], rsum[8];
#pragma unroll
  for (int r = 0; r < 8; ++r) { rmax[r] = -INFINITY; rsum[r] = 0.0f; }

  for (int kt = 0; kt < SEQ / 32; ++kt) {
    // stage K tile [32 tok][64 d] and V^T tile [64 d][32 tok]
    {
      int tok = tid >> 2, dbase = (tid & 3) * 16;
      const bf16* krow = Kb + (base + kt * 32 + tok) * DMODEL;
#pragma unroll
      for (int i = 0; i < 16; ++i)
        sK[tok][dbase + i] = krow[(dbase + i) * NHEADS + h];
      int d = tid >> 1, tbase = (tid & 1) * 16;
#pragma unroll
      for (int i = 0; i < 16; ++i)
        sVt[d][tbase + i] = Vb[(base + kt * 32 + tbase + i) * DMODEL + d * NHEADS + h];
    }
    __syncthreads();

    // S = Q @ K^T for 32 columns (two 16-col WMMA chunks, K-dim 64)
    v8f s0 = v8f{0}, s1 = v8f{0};
#pragma unroll
    for (int c = 0; c < 2; ++c) {
      Frag bk;
      bk.q[0] = *(const uint4*)&sK[m_f][c * 32 + koff];
      bk.q[1] = *(const uint4*)&sK[m_f][c * 32 + 16 + koff];
      s0 = __builtin_amdgcn_wmma_f32_16x16x32_bf16(false, qf[c].v, false, bk.v,
                                                   (short)0, s0, false, false);
      Frag bk1;
      bk1.q[0] = *(const uint4*)&sK[16 + m_f][c * 32 + koff];
      bk1.q[1] = *(const uint4*)&sK[16 + m_f][c * 32 + 16 + koff];
      s1 = __builtin_amdgcn_wmma_f32_16x16x32_bf16(false, qf[c].v, false, bk1.v,
                                                   (short)0, s1, false, false);
    }

    // online softmax (rows live within a 16-lane half)
#pragma unroll
    for (int r = 0; r < 8; ++r) {
      float a0 = s0[r] * inv_scale;
      float a1 = s1[r] * inv_scale;
      float mnew = half_max(fmaxf(fmaxf(a0, a1), rmax[r]));
      float alpha = __expf(rmax[r] - mnew);
      rmax[r] = mnew;
      float p0 = __expf(a0 - mnew);
      float p1 = __expf(a1 - mnew);
      rsum[r] = rsum[r] * alpha + half_sum(p0 + p1);
#pragma unroll
      for (int j = 0; j < 4; ++j) oacc[j][r] *= alpha;
      sP[wave][mbase + r][nloc]      = (bf16)p0;
      sP[wave][mbase + r][16 + nloc] = (bf16)p1;
    }
    __syncthreads();

    // O += P @ V   (A = P 16x32, B = V 32x16 per 16-d column tile)
    Frag pf;
    pf.q[0] = *(const uint4*)&sP[wave][m_f][koff];
    pf.q[1] = *(const uint4*)&sP[wave][m_f][16 + koff];
#pragma unroll
    for (int j = 0; j < 4; ++j) {
      Frag bv;
      bv.q[0] = *(const uint4*)&sVt[j * 16 + m_f][koff];
      bv.q[1] = *(const uint4*)&sVt[j * 16 + m_f][16 + koff];
      oacc[j] = __builtin_amdgcn_wmma_f32_16x16x32_bf16(false, pf.v, false, bv.v,
                                                        (short)0, oacc[j], false, false);
    }
    __syncthreads();
  }

  // write O, channel = h*DPH + d
#pragma unroll
  for (int j = 0; j < 4; ++j)
#pragma unroll
    for (int r = 0; r < 8; ++r) {
      float v = oacc[j][r] / rsum[r];
      size_t tokg = base + qblk * 64 + wave * 16 + mbase + r;
      Ob[tokg * DMODEL + h * DPH + j * 16 + nloc] = (bf16)v;
    }
}

// ---------------- LayerNorm (per row of DMODEL) ----------------
__global__ __launch_bounds__(256) void layernorm_k(
    const float* __restrict__ in, const float* __restrict__ g,
    const float* __restrict__ bt, bf16* __restrict__ outB,
    float* __restrict__ outF) {
  const int row = blockIdx.x;
  const float* x = in + (size_t)row * DMODEL;
  __shared__ float red[256];
  const int tid = threadIdx.x;

  float s = 0.0f;
  for (int i = tid; i < DMODEL; i += 256) s += x[i];
  red[tid] = s; __syncthreads();
  for (int off = 128; off > 0; off >>= 1) {
    if (tid < off) red[tid] += red[tid + off];
    __syncthreads();
  }
  float mu = red[0] * (1.0f / DMODEL);
  __syncthreads();

  float v = 0.0f;
  for (int i = tid; i < DMODEL; i += 256) { float d = x[i] - mu; v += d * d; }
  red[tid] = v; __syncthreads();
  for (int off = 128; off > 0; off >>= 1) {
    if (tid < off) red[tid] += red[tid + off];
    __syncthreads();
  }
  float rstd = rsqrtf(red[0] * (1.0f / DMODEL) + 1e-5f);

  for (int i = tid; i < DMODEL; i += 256) {
    float y = (x[i] - mu) * rstd * g[i] + bt[i];
    if (outB) outB[(size_t)row * DMODEL + i] = (bf16)y;
    if (outF) outF[(size_t)row * DMODEL + i] = y;
  }
}

// ---------------- host orchestration ----------------
extern "C" void kernel_launch(void* const* d_in, const int* in_sizes, int n_in,
                              void* d_out, int out_size, void* d_ws, size_t ws_size,
                              hipStream_t stream) {
  const float* x      = (const float*)d_in[0];
  const float* w_qvk  = (const float*)d_in[1];
  const float* b_qvk  = (const float*)d_in[2];
  const float* w_q    = (const float*)d_in[3];
  const float* b_q    = (const float*)d_in[4];
  const float* w_k    = (const float*)d_in[5];
  const float* b_k    = (const float*)d_in[6];
  const float* w_v    = (const float*)d_in[7];
  const float* b_v    = (const float*)d_in[8];
  const float* w_o    = (const float*)d_in[9];
  const float* b_o    = (const float*)d_in[10];
  const float* ln1_g  = (const float*)d_in[11];
  const float* ln1_b  = (const float*)d_in[12];
  const float* w_m1   = (const float*)d_in[13];
  const float* b_m1   = (const float*)d_in[14];
  const float* w_m2   = (const float*)d_in[15];
  const float* b_m2   = (const float*)d_in[16];
  const float* ln2_g  = (const float*)d_in[17];
  const float* ln2_b  = (const float*)d_in[18];
  float* out = (float*)d_out;

  char* ws = (char*)d_ws;
  size_t off = 0;
  auto alloc = [&](size_t bytes) { size_t o = off; off += (bytes + 255) & ~(size_t)255; return o; };

  const size_t o_xb    = alloc((size_t)TOK * DMODEL * 2);
  const size_t o_wqvkT = alloc((size_t)DQVK * DMODEL * 2);
  const size_t o_wqT   = alloc((size_t)DMODEL * DMODEL * 2);
  const size_t o_wkT   = alloc((size_t)DMODEL * DMODEL * 2);
  const size_t o_wvT   = alloc((size_t)DMODEL * DMODEL * 2);
  const size_t o_woT   = alloc((size_t)DMODEL * DMODEL * 2);
  const size_t o_wm1T  = alloc((size_t)DE * DMODEL * 2);
  const size_t o_wm2T  = alloc((size_t)DMODEL * DE * 2);
  const size_t o_qvkb  = alloc((size_t)TOK * DQVK * 2);   // later reused as hmid (bf16)
  const size_t o_qb    = alloc((size_t)TOK * DMODEL * 2); // qb+kb later reused as fp32 temp
  const size_t o_kb    = alloc((size_t)TOK * DMODEL * 2);
  const size_t o_vb    = alloc((size_t)TOK * DMODEL * 2); // later reused as post-LN1 bf16
  const size_t o_vhb   = alloc((size_t)TOK * DMODEL * 2);
  (void)ws_size; (void)in_sizes; (void)n_in; (void)out_size;

  bf16* xb    = (bf16*)(ws + o_xb);
  bf16* wqvkT = (bf16*)(ws + o_wqvkT);
  bf16* wqT   = (bf16*)(ws + o_wqT);
  bf16* wkT   = (bf16*)(ws + o_wkT);
  bf16* wvT   = (bf16*)(ws + o_wvT);
  bf16* woT   = (bf16*)(ws + o_woT);
  bf16* wm1T  = (bf16*)(ws + o_wm1T);
  bf16* wm2T  = (bf16*)(ws + o_wm2T);
  bf16* qvkb  = (bf16*)(ws + o_qvkb);
  bf16* qb    = (bf16*)(ws + o_qb);
  bf16* kb    = (bf16*)(ws + o_kb);
  bf16* vb    = (bf16*)(ws + o_vb);
  bf16* vhb   = (bf16*)(ws + o_vhb);
  bf16*  hmid  = qvkb;              // reuse (dead after Q/K/V projections)
  float* tmpF  = (float*)qb;        // reuse qb+kb (dead after attention)
  bf16*  attnb = vb;                // reuse vb (dead after attention)

  auto cdiv = [](size_t a, size_t b) { return (unsigned)((a + b - 1) / b); };

  // converts
  cvt_bf16<<<cdiv((size_t)TOK * DMODEL, 256), 256, 0, stream>>>(x, xb, (size_t)TOK * DMODEL);
  cvt_transpose_bf16<<<cdiv((size_t)DMODEL * DQVK, 256), 256, 0, stream>>>(w_qvk, wqvkT, DMODEL, DQVK);
  cvt_transpose_bf16<<<cdiv((size_t)DMODEL * DMODEL, 256), 256, 0, stream>>>(w_q, wqT, DMODEL, DMODEL);
  cvt_transpose_bf16<<<cdiv((size_t)DMODEL * DMODEL, 256), 256, 0, stream>>>(w_k, wkT, DMODEL, DMODEL);
  cvt_transpose_bf16<<<cdiv((size_t)DMODEL * DMODEL, 256), 256, 0, stream>>>(w_v, wvT, DMODEL, DMODEL);
  cvt_transpose_bf16<<<cdiv((size_t)DMODEL * DMODEL, 256), 256, 0, stream>>>(w_o, woT, DMODEL, DMODEL);
  cvt_transpose_bf16<<<cdiv((size_t)DMODEL * DE, 256), 256, 0, stream>>>(w_m1, wm1T, DMODEL, DE);
  cvt_transpose_bf16<<<cdiv((size_t)DE * DMODEL, 256), 256, 0, stream>>>(w_m2, wm2T, DE, DMODEL);

  // qvk = relu6(x @ w_qvk + b_qvk)
  gemm_bias_relu6<<<dim3(DQVK / 128, TOK / 128), 256, GSMEM_BYTES, stream>>>(
      xb, DMODEL, wqvkT, b_qvk, nullptr, qvkb, TOK, DQVK, DMODEL);
  // Q/K/V projections from qvk column slices (lda = DQVK)
  gemm_bias_relu6<<<dim3(DMODEL / 128, TOK / 128), 256, GSMEM_BYTES, stream>>>(
      qvkb + 0,          DQVK, wqT, b_q, nullptr, qb, TOK, DMODEL, DMODEL);
  gemm_bias_relu6<<<dim3(DMODEL / 128, TOK / 128), 256, GSMEM_BYTES, stream>>>(
      qvkb + DMODEL,     DQVK, wkT, b_k, nullptr, kb, TOK, DMODEL, DMODEL);
  gemm_bias_relu6<<<dim3(DMODEL / 128, TOK / 128), 256, GSMEM_BYTES, stream>>>(
      qvkb + 2 * DMODEL, DQVK, wvT, b_v, nullptr, vb, TOK, DMODEL, DMODEL);

  // attention
  const float inv_scale = 1.0f / (sqrtf((float)DPH) + 1e-10f);
  flash_attn<<<dim3(SEQ / 64, NHEADS, BATCH), 128, 0, stream>>>(qb, kb, vb, vhb, inv_scale);

  // attout = relu6(vheads @ w_o + b_o) -> fp32, then LN1 -> bf16
  gemm_bias_relu6<<<dim3(DMODEL / 128, TOK / 128), 256, GSMEM_BYTES, stream>>>(
      vhb, DMODEL, woT, b_o, tmpF, nullptr, TOK, DMODEL, DMODEL);
  layernorm_k<<<TOK, 256, 0, stream>>>(tmpF, ln1_g, ln1_b, attnb, nullptr);

  // MLP
  gemm_bias_relu6<<<dim3(DE / 128, TOK / 128), 256, GSMEM_BYTES, stream>>>(
      attnb, DMODEL, wm1T, b_m1, nullptr, hmid, TOK, DE, DMODEL);
  gemm_bias_relu6<<<dim3(DMODEL / 128, TOK / 128), 256, GSMEM_BYTES, stream>>>(
      hmid, DE, wm2T, b_m2, tmpF, nullptr, TOK, DMODEL, DE);
  layernorm_k<<<TOK, 256, 0, stream>>>(tmpF, ln2_g, ln2_b, nullptr, out);
}
